// MoELayer_31078383354369
// MI455X (gfx1250) — matrix operand order repaired and checked by reading, not compiled, not run on hardware
//
#include <hip/hip_runtime.h>
#include <hip/hip_bf16.h>

// MoE top-2 dispatch for MI455X (gfx1250), wave32, WMMA bf16.
// T=4096 tokens, D=1024, H=2048, E=8, top-k=2.
// Sparse per-expert grouped GEMMs, double-buffered LDS, 1 barrier/K-step.

#define T_TOK 4096
#define DDIM  1024
#define HDIM  2048
#define EEXP  8

typedef __attribute__((ext_vector_type(16))) __bf16 v16bf;
typedef __attribute__((ext_vector_type(8)))  float  v8f;

// ---------- helpers ----------

// native fp32 -> bf16 (fptrunc, round-to-nearest-even; lowers to packed
// v_cvt_*bf16 on gfx1250 instead of a bfe/add3/mov bit-twiddle sequence)
__device__ __forceinline__ unsigned short f2bf(float f) {
  __bf16 b = (__bf16)f;
  unsigned short s;
  __builtin_memcpy(&s, &b, 2);
  return s;
}

// convert 8 fp32 (two float4 in regs) -> 8 bf16, one 16B LDS store
__device__ __forceinline__ void cvt8r(float4 a, float4 b,
                                      unsigned short* __restrict__ dst) {
  alignas(16) __bf16 t[8] = { (__bf16)a.x, (__bf16)a.y, (__bf16)a.z, (__bf16)a.w,
                              (__bf16)b.x, (__bf16)b.y, (__bf16)b.z, (__bf16)b.w };
  *(int4*)dst = *(int4*)t;
}

// A fragment (16x32 bf16, MxK). ISA 7.12.2: lane<16 row=lane K in {0..7,16..23};
// lane>=16 row=lane-16, K in {8..15,24..31}. LDS row-major, strideE elements.
__device__ __forceinline__ v16bf frag_a(const unsigned short* base, int strideE,
                                        int row0, int lane) {
  const unsigned short* p = base + (row0 + (lane & 15)) * strideE + ((lane >> 4) << 3);
  int4 c0 = *(const int4*)p;
  int4 c1 = *(const int4*)(p + 16);
  v16bf f;
  __builtin_memcpy(&f, &c0, 16);
  __builtin_memcpy((char*)&f + 16, &c1, 16);
  return f;
}

// B fragment (32x16 bf16, KxN). Lanes 0-15: N=lane, K in {0..7,16..23}; lanes 16-31: K in {8..15,24..31}.
// LDS stored column-major: Bs[col][k], strideE elements per column.
__device__ __forceinline__ v16bf frag_b(const unsigned short* base, int strideE,
                                        int col0, int lane) {
  const unsigned short* p = base + (col0 + (lane & 15)) * strideE + ((lane >> 4) << 3);
  int4 c0 = *(const int4*)p;
  int4 c1 = *(const int4*)(p + 16);
  v16bf f;
  __builtin_memcpy(&f, &c0, 16);
  __builtin_memcpy((char*)&f + 16, &c1, 16);
  return f;
}

__device__ __forceinline__ v8f wmma_bf16(v16bf a, v16bf b, v8f c) {
  return __builtin_amdgcn_wmma_f32_16x16x32_bf16(false, a, false, b, (short)0, c,
                                                 false, false);
}

// ---------- kernel 0: zero out + counts ----------

__global__ __launch_bounds__(256) void zero_init(float* __restrict__ out, int n4,
                                                 int* __restrict__ counts) {
  int i = blockIdx.x * blockDim.x + threadIdx.x;
  if (i < n4) ((float4*)out)[i] = make_float4(0.f, 0.f, 0.f, 0.f);
  if (i < EEXP) counts[i] = 0;
}

// ---------- kernel 1: router logits (one wave per token) ----------

__global__ __launch_bounds__(256) void router_logits(const float* __restrict__ x,
                                                     const float* __restrict__ rw,
                                                     float* __restrict__ logits) {
  int t = blockIdx.x * 8 + (threadIdx.x >> 5);
  int lane = threadIdx.x & 31;
  const float* xr = x + (size_t)t * DDIM;
  float acc[EEXP];
#pragma unroll
  for (int e = 0; e < EEXP; e++) acc[e] = 0.f;
  for (int k = lane; k < DDIM; k += 32) {
    float xv = xr[k];
#pragma unroll
    for (int e = 0; e < EEXP; e++) acc[e] += xv * rw[e * DDIM + k];
  }
#pragma unroll
  for (int e = 0; e < EEXP; e++) {
    float v = acc[e];
#pragma unroll
    for (int m = 16; m > 0; m >>= 1) v += __shfl_xor(v, m, 32);
    if (lane == 0) logits[t * EEXP + e] = v;
  }
}

// ---------- kernel 2: top-2 + softmax + scatter to expert lists ----------

__global__ __launch_bounds__(256) void router_topk(const float* __restrict__ logits,
                                                   int* __restrict__ counts,
                                                   int* __restrict__ rows,
                                                   float* __restrict__ rwt) {
  int t = blockIdx.x * blockDim.x + threadIdx.x;
  if (t >= T_TOK) return;
  const float* lg = logits + t * EEXP;
  float b0 = -3.4e38f, b1 = -3.4e38f;
  int i0 = 0, i1 = 0;
#pragma unroll
  for (int e = 0; e < EEXP; e++) {
    float v = lg[e];
    if (v > b0) { b1 = b0; i1 = i0; b0 = v; i0 = e; }
    else if (v > b1) { b1 = v; i1 = e; }
  }
  float e1 = __expf(b1 - b0);
  float inv = 1.0f / (1.0f + e1);
  int p0 = atomicAdd(&counts[i0], 1);
  rows[i0 * T_TOK + p0] = t;
  rwt[i0 * T_TOK + p0] = inv;
  int p1 = atomicAdd(&counts[i1], 1);
  rows[i1 * T_TOK + p1] = t;
  rwt[i1 * T_TOK + p1] = e1 * inv;
}

// ---------- kernel 3: per-expert h = silu(x w1^T) * (x w3^T), bf16 out ----------
// Tile: M=128 gathered tokens, N=64 of H, K-step 32 over D. 8 waves = 4(M)x2(N).
// Double-buffered LDS: global loads for tile k+1 issue before the WMMAs of tile k.

#define LDSTRIDE 40  // 32 + 8 pad elements -> 80B rows, conflict-free b128 frag loads

__global__ __launch_bounds__(256) void moe_ffn1(const float* __restrict__ x,
                                                const float* __restrict__ w1,
                                                const float* __restrict__ w3,
                                                const int* __restrict__ counts,
                                                const int* __restrict__ rows,
                                                unsigned short* __restrict__ h,
                                                int e) {
  __shared__ unsigned short As[2][128 * LDSTRIDE];
  __shared__ unsigned short B1s[2][64 * LDSTRIDE];
  __shared__ unsigned short B3s[2][64 * LDSTRIDE];

  int cnt = counts[e];
  int mBase = blockIdx.x * 128;
  if (mBase >= cnt) return;
  int nBase = blockIdx.y * 64;

  const float* w1e = w1 + (size_t)e * HDIM * DDIM;
  const float* w3e = w3 + (size_t)e * HDIM * DDIM;
  const int* rowsE = rows + e * T_TOK;

  int tid = threadIdx.x, lane = tid & 31, wave = tid >> 5;
  int wm = wave & 3, wn = wave >> 2;

  // A staging: 2 threads per row, 16 fp32 each (gathered token row)
  int aRow = tid >> 1, aK0 = (tid & 1) * 16;
  int r = mBase + aRow;
  int aTok = (r < cnt) ? rowsE[r] : 0;
  const float* aSrc = x + (size_t)aTok * DDIM + aK0;
  // B staging: 4 threads per column, 8 fp32 each
  int bCol = tid >> 2, bK0 = (tid & 3) * 8;
  const float* b1Src = w1e + (size_t)(nBase + bCol) * DDIM + bK0;
  const float* b3Src = w3e + (size_t)(nBase + bCol) * DDIM + bK0;

  v8f z = {0.f, 0.f, 0.f, 0.f, 0.f, 0.f, 0.f, 0.f};
  v8f accG[2][2] = {{z, z}, {z, z}};
  v8f accU[2][2] = {{z, z}, {z, z}};

  float4 sa0, sa1, sa2, sa3, sb10, sb11, sb30, sb31;

  // prologue: stage tile 0 into buffer 0
  {
    const float4* pa = (const float4*)(aSrc);
    sa0 = pa[0]; sa1 = pa[1]; sa2 = pa[2]; sa3 = pa[3];
    const float4* p1 = (const float4*)(b1Src);
    sb10 = p1[0]; sb11 = p1[1];
    const float4* p3 = (const float4*)(b3Src);
    sb30 = p3[0]; sb31 = p3[1];
    cvt8r(sa0, sa1, &As[0][aRow * LDSTRIDE + aK0]);
    cvt8r(sa2, sa3, &As[0][aRow * LDSTRIDE + aK0 + 8]);
    cvt8r(sb10, sb11, &B1s[0][bCol * LDSTRIDE + bK0]);
    cvt8r(sb30, sb31, &B3s[0][bCol * LDSTRIDE + bK0]);
  }
  __syncthreads();

  for (int kb = 0; kb < DDIM; kb += 32) {
    int cur = (kb >> 5) & 1;
    int nxt = cur ^ 1;
    bool hasNext = (kb + 32) < DDIM;

    if (hasNext) {  // issue next tile's global loads early
      const float4* pa = (const float4*)(aSrc + kb + 32);
      sa0 = pa[0]; sa1 = pa[1]; sa2 = pa[2]; sa3 = pa[3];
      const float4* p1 = (const float4*)(b1Src + kb + 32);
      sb10 = p1[0]; sb11 = p1[1];
      const float4* p3 = (const float4*)(b3Src + kb + 32);
      sb30 = p3[0]; sb31 = p3[1];
    }

    v16bf a0 = frag_a(As[cur], LDSTRIDE, wm * 32,      lane);
    v16bf a1 = frag_a(As[cur], LDSTRIDE, wm * 32 + 16, lane);
    v16bf p0 = frag_b(B1s[cur], LDSTRIDE, wn * 32,      lane);
    v16bf p1 = frag_b(B1s[cur], LDSTRIDE, wn * 32 + 16, lane);
    v16bf q0 = frag_b(B3s[cur], LDSTRIDE, wn * 32,      lane);
    v16bf q1 = frag_b(B3s[cur], LDSTRIDE, wn * 32 + 16, lane);

    accG[0][0] = wmma_bf16(a0, p0, accG[0][0]);
    accG[0][1] = wmma_bf16(a0, p1, accG[0][1]);
    accG[1][0] = wmma_bf16(a1, p0, accG[1][0]);
    accG[1][1] = wmma_bf16(a1, p1, accG[1][1]);
    accU[0][0] = wmma_bf16(a0, q0, accU[0][0]);
    accU[0][1] = wmma_bf16(a0, q1, accU[0][1]);
    accU[1][0] = wmma_bf16(a1, q0, accU[1][0]);
    accU[1][1] = wmma_bf16(a1, q1, accU[1][1]);

    if (hasNext) {  // convert + park next tile in the alternate buffer
      cvt8r(sa0, sa1, &As[nxt][aRow * LDSTRIDE + aK0]);
      cvt8r(sa2, sa3, &As[nxt][aRow * LDSTRIDE + aK0 + 8]);
      cvt8r(sb10, sb11, &B1s[nxt][bCol * LDSTRIDE + bK0]);
      cvt8r(sb30, sb31, &B3s[nxt][bCol * LDSTRIDE + bK0]);
    }
    __syncthreads();
  }

  int hi = lane >> 4, nlo = lane & 15;
#pragma unroll
  for (int sm = 0; sm < 2; sm++) {
#pragma unroll
    for (int vr = 0; vr < 8; vr++) {
      int rr = mBase + wm * 32 + sm * 16 + vr + hi * 8;
      if (rr < cnt) {
#pragma unroll
        for (int sn = 0; sn < 2; sn++) {
          int col = nBase + wn * 32 + sn * 16 + nlo;
          float g = accG[sm][sn][vr];
          float u = accU[sm][sn][vr];
          float hv = __fdividef(g, 1.0f + __expf(-g)) * u;  // silu(g)*u
          h[(size_t)rr * HDIM + col] = f2bf(hv);
        }
      }
    }
  }
}

// ---------- kernel 4: y = h w2^T, atomic scatter-add out[t] += w * y ----------

__global__ __launch_bounds__(256) void moe_ffn2(const unsigned short* __restrict__ h,
                                                const float* __restrict__ w2,
                                                const int* __restrict__ counts,
                                                const int* __restrict__ rows,
                                                const float* __restrict__ rwt,
                                                float* __restrict__ out, int e) {
  __shared__ unsigned short As[2][128 * LDSTRIDE];
  __shared__ unsigned short Bs[2][64 * LDSTRIDE];

  int cnt = counts[e];
  int mBase = blockIdx.x * 128;
  if (mBase >= cnt) return;
  int nBase = blockIdx.y * 64;

  const float* w2e = w2 + (size_t)e * DDIM * HDIM;
  const int* rowsE = rows + e * T_TOK;
  const float* rwtE = rwt + e * T_TOK;

  int tid = threadIdx.x, lane = tid & 31, wave = tid >> 5;
  int wm = wave & 3, wn = wave >> 2;

  int aRow = tid >> 1, aK0 = (tid & 1) * 16;
  const unsigned short* aSrc = h + (size_t)(mBase + aRow) * HDIM + aK0;
  int bCol = tid >> 2, bK0 = (tid & 3) * 8;
  const float* bSrc = w2e + (size_t)(nBase + bCol) * HDIM + bK0;

  v8f z = {0.f, 0.f, 0.f, 0.f, 0.f, 0.f, 0.f, 0.f};
  v8f acc[2][2] = {{z, z}, {z, z}};

  uint4 ha0, ha1;
  float4 wb0, wb1;

  {  // prologue: stage tile 0
    const uint4* s = (const uint4*)(aSrc);
    ha0 = s[0]; ha1 = s[1];
    const float4* p = (const float4*)(bSrc);
    wb0 = p[0]; wb1 = p[1];
    *(uint4*)&As[0][aRow * LDSTRIDE + aK0]     = ha0;
    *(uint4*)&As[0][aRow * LDSTRIDE + aK0 + 8] = ha1;
    cvt8r(wb0, wb1, &Bs[0][bCol * LDSTRIDE + bK0]);
  }
  __syncthreads();

  for (int kb = 0; kb < HDIM; kb += 32) {
    int cur = (kb >> 5) & 1;
    int nxt = cur ^ 1;
    bool hasNext = (kb + 32) < HDIM;

    if (hasNext) {
      const uint4* s = (const uint4*)(aSrc + kb + 32);
      ha0 = s[0]; ha1 = s[1];
      const float4* p = (const float4*)(bSrc + kb + 32);
      wb0 = p[0]; wb1 = p[1];
    }

    v16bf a0 = frag_a(As[cur], LDSTRIDE, wm * 32,      lane);
    v16bf a1 = frag_a(As[cur], LDSTRIDE, wm * 32 + 16, lane);
    v16bf b0 = frag_b(Bs[cur], LDSTRIDE, wn * 32,      lane);
    v16bf b1 = frag_b(Bs[cur], LDSTRIDE, wn * 32 + 16, lane);

    acc[0][0] = wmma_bf16(a0, b0, acc[0][0]);
    acc[0][1] = wmma_bf16(a0, b1, acc[0][1]);
    acc[1][0] = wmma_bf16(a1, b0, acc[1][0]);
    acc[1][1] = wmma_bf16(a1, b1, acc[1][1]);

    if (hasNext) {
      *(uint4*)&As[nxt][aRow * LDSTRIDE + aK0]     = ha0;
      *(uint4*)&As[nxt][aRow * LDSTRIDE + aK0 + 8] = ha1;
      cvt8r(wb0, wb1, &Bs[nxt][bCol * LDSTRIDE + bK0]);
    }
    __syncthreads();
  }

  int hi = lane >> 4, nlo = lane & 15;
#pragma unroll
  for (int sm = 0; sm < 2; sm++) {
#pragma unroll
    for (int vr = 0; vr < 8; vr++) {
      int rr = mBase + wm * 32 + sm * 16 + vr + hi * 8;
      if (rr < cnt) {
        int t = rowsE[rr];
        float w = rwtE[rr];
#pragma unroll
        for (int sn = 0; sn < 2; sn++) {
          int col = nBase + wn * 32 + sn * 16 + nlo;
          // one atomic per (t,col) per expert launch; launches serialize on the
          // stream, so the result is deterministic.
          unsafeAtomicAdd(&out[(size_t)t * DDIM + col], w * acc[sm][sn][vr]);
        }
      }
    }
  }
}

// ---------- launcher ----------

extern "C" void kernel_launch(void* const* d_in, const int* in_sizes, int n_in,
                              void* d_out, int out_size, void* d_ws, size_t ws_size,
                              hipStream_t stream) {
  const float* x  = (const float*)d_in[0];  // [2,2048,1024]
  const float* rw = (const float*)d_in[1];  // [8,1024]
  const float* w1 = (const float*)d_in[2];  // [8,2048,1024]
  const float* w2 = (const float*)d_in[3];  // [8,1024,2048]
  const float* w3 = (const float*)d_in[4];  // [8,2048,1024]
  float* out = (float*)d_out;               // [2,2048,1024]

  char* ws = (char*)d_ws;
  float* logits        = (float*)(ws);                     // T*E*4   = 131072
  int*   counts        = (int*)(ws + 131072);              // 256 (E ints, padded)
  int*   rows          = (int*)(ws + 131328);              // E*T*4   = 131072
  float* rwt           = (float*)(ws + 262400);            // E*T*4   = 131072
  unsigned short* hbuf = (unsigned short*)(ws + 393472);   // T*H*2   = 16.8MB

  zero_init<<<dim3((T_TOK * DDIM / 4 + 255) / 256), dim3(256), 0, stream>>>(
      out, T_TOK * DDIM / 4, counts);
  router_logits<<<dim3(T_TOK / 8), dim3(256), 0, stream>>>(x, rw, logits);
  router_topk<<<dim3(T_TOK / 256), dim3(256), 0, stream>>>(logits, counts, rows, rwt);

  for (int e = 0; e < EEXP; e++) {
    moe_ffn1<<<dim3(T_TOK / 128, HDIM / 64), dim3(256), 0, stream>>>(
        x, w1, w3, counts, rows, hbuf, e);
    moe_ffn2<<<dim3(T_TOK / 128, DDIM / 64), dim3(256), 0, stream>>>(
        hbuf, w2, counts, rows, rwt, out, e);
  }
}
